// CausalSelfAttention_9826885173974
// MI455X (gfx1250) — matrix operand order repaired and checked by reading, not compiled, not run on hardware
//
#include <hip/hip_runtime.h>
#include <hip/hip_bf16.h>

#define B_  2
#define T_  2048
#define C_  1024
#define H_  16
#define HD_ 64
#define M_  (B_*T_)   // 4096 rows for the projections

typedef __bf16 bf16_t;
typedef __attribute__((ext_vector_type(16))) __bf16 v16bf;
typedef __attribute__((ext_vector_type(8)))  __bf16 v8bf;
typedef __attribute__((ext_vector_type(8)))  float  v8f;

// ---------- helpers ----------

__device__ inline bf16_t f2bf(float f) {
  union { float f; unsigned u; } v; v.f = f;
  unsigned r = v.u + 0x7FFFu + ((v.u >> 16) & 1u);   // round-to-nearest-even
  unsigned short h = (unsigned short)(r >> 16);
  bf16_t o; __builtin_memcpy(&o, &h, 2);
  return o;
}

#define SHUF16(lo, hi) __builtin_shufflevector(lo, hi, 0,1,2,3,4,5,6,7,8,9,10,11,12,13,14,15)

// A-fragment (16x32, bf16), row-major source with leading dim ld.
// Layout used here (contiguous k, same as B): lane row m = lane%16; elems i=0..15 -> k = col0 + 16*half + i.
__device__ inline v16bf load_a_frag(const bf16_t* p, int ld, int row0, int col0,
                                    int half, int l16) {
  const bf16_t* base = p + (size_t)(row0 + l16) * ld + col0 + 16 * half;
  v8bf lo = *(const v8bf*)base;          // k = col0+16*half .. +7
  v8bf hi = *(const v8bf*)(base + 8);    // k = col0+16*half+8 .. +15
  return SHUF16(lo, hi);
}

// B-fragment (32x16, bf16) where B[k][n] = src[n][k] (src row-major [N,K], ld).
// Layout used here (contiguous k, same as A): lane col n = lane%16; elems i=0..15 -> k = k0 + 16*half + i.
__device__ inline v16bf load_bt_frag(const bf16_t* p, int ld, int n0, int k0,
                                     int half, int l16) {
  const bf16_t* base = p + (size_t)(n0 + l16) * ld + k0 + 16 * half;
  v8bf lo = *(const v8bf*)base;
  v8bf hi = *(const v8bf*)(base + 8);
  return SHUF16(lo, hi);
}

__device__ inline v8f wmma_bf(v16bf a, v16bf b, v8f c) {
  return __builtin_amdgcn_wmma_f32_16x16x32_bf16(false, a, false, b,
                                                 (short)0, c, false, false);
}

// Row of the C/D tile lives in 16 lanes (0-15 or 16-31); reduce across them.
__device__ inline float redmax16(float v) {
  v = fmaxf(v, __shfl_xor(v, 1, 32));
  v = fmaxf(v, __shfl_xor(v, 2, 32));
  v = fmaxf(v, __shfl_xor(v, 4, 32));
  v = fmaxf(v, __shfl_xor(v, 8, 32));
  return v;
}
__device__ inline float redsum16(float v) {
  v += __shfl_xor(v, 1, 32);
  v += __shfl_xor(v, 2, 32);
  v += __shfl_xor(v, 4, 32);
  v += __shfl_xor(v, 8, 32);
  return v;
}

// ---------- kernels ----------

__global__ void cvt_bf16_kernel(const float* __restrict__ in,
                                bf16_t* __restrict__ out, int n) {
  int i = blockIdx.x * blockDim.x + threadIdx.x;
  int stride = gridDim.x * blockDim.x;
  for (; i < n; i += stride) out[i] = f2bf(in[i]);
}

// y[m,n] = sum_k A[m,k]*W[n,k] + bias[n]; bf16 out either as [B,H,T,64] (Q/K)
// or V-transposed [B,H,64,T]. Wave tile: 32(M) x 64(N). Block: 8 waves = 256 M-rows.
__global__ __launch_bounds__(256) void qkv_gemm_kernel(
    const bf16_t* __restrict__ A, const bf16_t* __restrict__ W,
    const float* __restrict__ bias, bf16_t* __restrict__ out, int transposedV) {
  const int wave = threadIdx.x >> 5, lane = threadIdx.x & 31;
  const int half = lane >> 4, l16 = lane & 15;
  const int m0 = blockIdx.x * 256 + wave * 32;
  const int n0 = blockIdx.y * 64;

  v8f acc[2][4];
#pragma unroll
  for (int i = 0; i < 2; ++i)
#pragma unroll
    for (int j = 0; j < 4; ++j) acc[i][j] = (v8f){0,0,0,0,0,0,0,0};

  for (int kb = 0; kb < C_; kb += 32) {
    v16bf a0 = load_a_frag(A, C_, m0,      kb, half, l16);
    v16bf a1 = load_a_frag(A, C_, m0 + 16, kb, half, l16);
#pragma unroll
    for (int ni = 0; ni < 4; ++ni) {
      v16bf b = load_bt_frag(W, C_, n0 + ni * 16, kb, half, l16);
      acc[0][ni] = wmma_bf(a0, b, acc[0][ni]);
      acc[1][ni] = wmma_bf(a1, b, acc[1][ni]);
    }
  }

#pragma unroll
  for (int ni = 0; ni < 4; ++ni) {
    const int n = n0 + ni * 16 + l16;
    const float bs = bias[n];
    const int h = n >> 6, d = n & (HD_ - 1);
#pragma unroll
    for (int mi = 0; mi < 2; ++mi) {
#pragma unroll
      for (int r = 0; r < 8; ++r) {
        const int m = m0 + mi * 16 + r + 8 * half;
        const int b = m >> 11, t = m & (T_ - 1);
        const float y = acc[mi][ni][r] + bs;
        size_t off;
        if (transposedV) off = ((size_t)(b * H_ + h) * HD_ + d) * T_ + t;
        else             off = ((size_t)(b * H_ + h) * T_  + t) * HD_ + d;
        out[off] = f2bf(y);
      }
    }
  }
}

// Flash attention: one wave per 16 q-rows, full HD=64. Q,K in [B,H,T,64],
// V transposed [B,H,64,T]. Output bf16 [B,T,C].
__global__ __launch_bounds__(256) void attn_kernel(
    const bf16_t* __restrict__ Q, const bf16_t* __restrict__ K,
    const bf16_t* __restrict__ Vt, bf16_t* __restrict__ O) {
  __shared__ __align__(16) bf16_t pbuf[8][16 * 32];   // per-wave P transpose buffer

  const int wave = threadIdx.x >> 5, lane = threadIdx.x & 31;
  const int half = lane >> 4, l16 = lane & 15;
  const int bh = blockIdx.y;
  const int b = bh / H_, h = bh % H_;
  const int qbase = blockIdx.x * 128 + wave * 16;

  const bf16_t* Qh = Q  + (size_t)bh * T_ * HD_;
  const bf16_t* Kh = K  + (size_t)bh * T_ * HD_;
  const bf16_t* Vh = Vt + (size_t)bh * HD_ * T_;

  const v16bf qa0 = load_a_frag(Qh, HD_, qbase,  0, half, l16);
  const v16bf qa1 = load_a_frag(Qh, HD_, qbase, 32, half, l16);

  v8f acc[4];
  float mi[8], li[8];
#pragma unroll
  for (int dt = 0; dt < 4; ++dt) acc[dt] = (v8f){0,0,0,0,0,0,0,0};
#pragma unroll
  for (int r = 0; r < 8; ++r) { mi[r] = -__builtin_inff(); li[r] = 0.0f; }

  const float scale = 0.125f;  // 1/sqrt(64)
  const int kend = qbase + 16; // causal: columns [0, qbase+16)

  for (int kb = 0; kb < kend; kb += 32) {
    // S = Q * K^T for 32 key columns (two 16x16 tiles, K-dim 64 = 2 WMMAs each)
    v8f s0 = (v8f){0,0,0,0,0,0,0,0}, s1 = (v8f){0,0,0,0,0,0,0,0};
    s0 = wmma_bf(qa0, load_bt_frag(Kh, HD_, kb,       0, half, l16), s0);
    s0 = wmma_bf(qa1, load_bt_frag(Kh, HD_, kb,      32, half, l16), s0);
    s1 = wmma_bf(qa0, load_bt_frag(Kh, HD_, kb + 16,  0, half, l16), s1);
    s1 = wmma_bf(qa1, load_bt_frag(Kh, HD_, kb + 16, 32, half, l16), s1);

    const int kc0 = kb + l16, kc1 = kb + 16 + l16;
#pragma unroll
    for (int r = 0; r < 8; ++r) {
      const int qg = qbase + r + 8 * half;
      float a = (kc0 <= qg) ? s0[r] * scale : -__builtin_inff();
      float c = (kc1 <= qg) ? s1[r] * scale : -__builtin_inff();
      const float rm   = redmax16(fmaxf(a, c));
      const float mnew = fmaxf(mi[r], rm);              // always finite (col kb unmasked)
      const float alpha = __expf(mi[r] - mnew);         // first iter: exp(-inf)=0
      const float p0 = __expf(a - mnew);
      const float p1 = __expf(c - mnew);
      mi[r] = mnew;
      li[r] = li[r] * alpha + redsum16(p0 + p1);
#pragma unroll
      for (int dt = 0; dt < 4; ++dt) acc[dt][r] *= alpha;
      const int row = (r + 8 * half) * 32;
      pbuf[wave][row + l16]      = f2bf(p0);
      pbuf[wave][row + 16 + l16] = f2bf(p1);
    }

    // Intra-wave LDS ordering before reading P back in A-fragment layout.
    asm volatile("s_wait_dscnt 0x0" ::: "memory");
    const v16bf pa = load_a_frag(&pbuf[wave][0], 32, 0, 0, half, l16);

    // acc += P * V  (B[k][d] = Vt[d][k], contiguous in k)
#pragma unroll
    for (int dt = 0; dt < 4; ++dt) {
      const v16bf vb = load_bt_frag(Vh, T_, dt * 16, kb, half, l16);
      acc[dt] = wmma_bf(pa, vb, acc[dt]);
    }
  }

#pragma unroll
  for (int r = 0; r < 8; ++r) {
    const float inv = 1.0f / li[r];
    const int t = qbase + r + 8 * half;
    bf16_t* orow = O + ((size_t)b * T_ + t) * C_ + h * HD_;
#pragma unroll
    for (int dt = 0; dt < 4; ++dt)
      orow[dt * 16 + l16] = f2bf(acc[dt][r] * inv);
  }
}

// Output projection: fp32 result into d_out [B,T,C].
__global__ __launch_bounds__(256) void proj_gemm_kernel(
    const bf16_t* __restrict__ A, const bf16_t* __restrict__ W,
    const float* __restrict__ bias, float* __restrict__ out) {
  const int wave = threadIdx.x >> 5, lane = threadIdx.x & 31;
  const int half = lane >> 4, l16 = lane & 15;
  const int m0 = blockIdx.x * 256 + wave * 32;
  const int n0 = blockIdx.y * 64;

  v8f acc[2][4];
#pragma unroll
  for (int i = 0; i < 2; ++i)
#pragma unroll
    for (int j = 0; j < 4; ++j) acc[i][j] = (v8f){0,0,0,0,0,0,0,0};

  for (int kb = 0; kb < C_; kb += 32) {
    v16bf a0 = load_a_frag(A, C_, m0,      kb, half, l16);
    v16bf a1 = load_a_frag(A, C_, m0 + 16, kb, half, l16);
#pragma unroll
    for (int ni = 0; ni < 4; ++ni) {
      v16bf b = load_bt_frag(W, C_, n0 + ni * 16, kb, half, l16);
      acc[0][ni] = wmma_bf(a0, b, acc[0][ni]);
      acc[1][ni] = wmma_bf(a1, b, acc[1][ni]);
    }
  }

#pragma unroll
  for (int ni = 0; ni < 4; ++ni) {
    const int n = n0 + ni * 16 + l16;
    const float bs = bias[n];
#pragma unroll
    for (int mi = 0; mi < 2; ++mi) {
#pragma unroll
      for (int r = 0; r < 8; ++r) {
        const int m = m0 + mi * 16 + r + 8 * half;
        out[(size_t)m * C_ + n] = acc[mi][ni][r] + bs;
      }
    }
  }
}

// ---------- launch ----------

extern "C" void kernel_launch(void* const* d_in, const int* in_sizes, int n_in,
                              void* d_out, int out_size, void* d_ws, size_t ws_size,
                              hipStream_t stream) {
  const float* x  = (const float*)d_in[0];
  const float* Wq = (const float*)d_in[1];
  const float* bq = (const float*)d_in[2];
  const float* Wk = (const float*)d_in[3];
  const float* bk = (const float*)d_in[4];
  const float* Wv = (const float*)d_in[5];
  const float* bv = (const float*)d_in[6];
  const float* Wp = (const float*)d_in[7];
  const float* bp = (const float*)d_in[8];

  const size_t nBTC = (size_t)B_ * T_ * C_;  // 4,194,304
  const size_t nCC  = (size_t)C_ * C_;       // 1,048,576

  bf16_t* ws  = (bf16_t*)d_ws;
  size_t  o   = 0;
  bf16_t* xb  = ws + o; o += nBTC;
  bf16_t* wqb = ws + o; o += nCC;
  bf16_t* wkb = ws + o; o += nCC;
  bf16_t* wvb = ws + o; o += nCC;
  bf16_t* wpb = ws + o; o += nCC;
  bf16_t* Qb  = ws + o; o += nBTC;   // [B,H,T,64]
  bf16_t* Kb  = ws + o; o += nBTC;   // [B,H,T,64]
  bf16_t* Vtb = ws + o; o += nBTC;   // [B,H,64,T]
  bf16_t* Ab  = ws + o; o += nBTC;   // attn out, [B,T,C]

  cvt_bf16_kernel<<<1024, 256, 0, stream>>>(x,  xb,  (int)nBTC);
  cvt_bf16_kernel<<<512,  256, 0, stream>>>(Wq, wqb, (int)nCC);
  cvt_bf16_kernel<<<512,  256, 0, stream>>>(Wk, wkb, (int)nCC);
  cvt_bf16_kernel<<<512,  256, 0, stream>>>(Wv, wvb, (int)nCC);
  cvt_bf16_kernel<<<512,  256, 0, stream>>>(Wp, wpb, (int)nCC);

  dim3 g(M_ / 256, C_ / 64);  // 16 x 16
  qkv_gemm_kernel<<<g, 256, 0, stream>>>(xb, wqb, bq, Qb,  0);
  qkv_gemm_kernel<<<g, 256, 0, stream>>>(xb, wkb, bk, Kb,  0);
  qkv_gemm_kernel<<<g, 256, 0, stream>>>(xb, wvb, bv, Vtb, 1);

  attn_kernel<<<dim3(T_ / 128, B_ * H_), 256, 0, stream>>>(Qb, Kb, Vtb, Ab);

  proj_gemm_kernel<<<g, 256, 0, stream>>>(Ab, wpb, bp, (float*)d_out);

  (void)in_sizes; (void)n_in; (void)out_size; (void)ws_size;
}
